// GRUContext_72834055406356
// MI455X (gfx1250) — compile-verified, matrix operand order checked by reading, not run on hardware
//
#include <hip/hip_runtime.h>
#include <stdint.h>

#define B_  64
#define T_  512
#define I_  512
#define H_  1024
#define G3H (3 * H_)

typedef __bf16 v16bf __attribute__((ext_vector_type(16)));
typedef __bf16 v8bf  __attribute__((ext_vector_type(8)));
typedef float  v8f   __attribute__((ext_vector_type(8)));

#define SHUF16(lo, hi) __builtin_shufflevector((lo), (hi), 0,1,2,3,4,5,6,7,8,9,10,11,12,13,14,15)

// ---- gfx1250 async global->LDS path (ASYNCcnt). Copy via inline asm (portable);
// ---- wait via the builtin that type-checks on this toolchain.
#ifndef TRY_ASYNC
#define TRY_ASYNC 1
#endif
#if TRY_ASYNC && defined(__AMDGCN__)
#  if __has_builtin(__builtin_amdgcn_s_wait_asynccnt)
#    define USE_ASYNC 1
#  endif
#endif
#ifndef USE_ASYNC
#define USE_ASYNC 0
#endif

#if USE_ASYNC
// Per-lane 16-byte async copy: global -> LDS. VDST carries the LDS byte offset
// (= low 32 bits of the generic pointer per the aperture rules); VADDR is the
// 64-bit global address; 'off' = no SADDR (GV mode).
__device__ __forceinline__ void async_cp16(const void* gsrc, const void* ldst) {
    uint32_t lds_off = (uint32_t)(uintptr_t)ldst;
    asm volatile("global_load_async_to_lds_b128 %0, %1, off"
                 :
                 : "v"(lds_off), "v"(gsrc)
                 : "memory");
}
#define ASYNC_CP16(gsrc, ldst) async_cp16((const void*)(gsrc), (const void*)(ldst))
#define ASYNC_WAIT() __builtin_amdgcn_s_wait_asynccnt(0)
#endif

// ---- scheduler grouping: batch DS/VMEM reads ahead of the WMMA block so the
// ---- fragments get distinct registers and the WMMAs issue back-to-back.
#if defined(__AMDGCN__)
#  if __has_builtin(__builtin_amdgcn_sched_group_barrier)
#    define HAVE_SGB 1
#  endif
#endif
#ifndef HAVE_SGB
#define HAVE_SGB 0
#endif
// masks: 0x008 = MFMA/WMMA, 0x020 = VMEM read, 0x100 = DS read
#if HAVE_SGB
#define SGB(mask, size) __builtin_amdgcn_sched_group_barrier((mask), (size), 0)
#else
#define SGB(mask, size)
#endif

__device__ __forceinline__ float sigmoid_f(float x) { return 1.0f / (1.0f + __expf(-x)); }

// ---------------- fp32 -> bf16 conversion (grid-stride) ----------------
__global__ void cvt_f32_bf16(const float* __restrict__ src, __bf16* __restrict__ dst, long long n) {
    long long i = (long long)blockIdx.x * blockDim.x + threadIdx.x;
    long long stride = (long long)gridDim.x * blockDim.x;
    for (; i < n; i += stride) dst[i] = (__bf16)src[i];
}

__global__ void init_barrier(unsigned* bar) { bar[0] = 0u; }

// ---------------- big GEMM: xg[t*B + b][3H] = x @ w_ih^T + b_ih ----------------
// A:  [M = B*T][K] bf16 row-major (m = b*T + t)
// Bw: [3H][K]     bf16 row-major
// out xg stored [t][b][3H] bf16. Double-buffered LDS; async staging when available.
__global__ __launch_bounds__(256) void gemm_xg(const __bf16* __restrict__ A,
                                               const __bf16* __restrict__ Bw,
                                               const float* __restrict__ bias,
                                               __bf16* __restrict__ xg, int K) {
    __shared__ __align__(16) __bf16 As[2][128 * 32];
    __shared__ __align__(16) __bf16 Bs[2][128 * 32];
    const int tid  = threadIdx.x;
    const int lane = tid & 31;
    const int wave = tid >> 5;        // 8 waves: each owns a 16-row M strip
    const int l16  = lane & 15;
    const int kch  = lane >> 4;       // 0/1: which 8-elem K chunk (ISA A/B layout)
    const int mbase = blockIdx.y * 128;
    const int nbase = blockIdx.x * 128;

    v8f acc[8];
#pragma unroll
    for (int i = 0; i < 8; ++i) {
        v8f z = {0.f, 0.f, 0.f, 0.f, 0.f, 0.f, 0.f, 0.f};
        acc[i] = z;
    }

    const int ldr = tid >> 1;         // 0..127: row staged by this thread
    const int ldc = (tid & 1) * 16;   // 0/16: K offset
    const __bf16* gA = A  + (size_t)(mbase + ldr) * K + ldc;
    const __bf16* gB = Bw + (size_t)(nbase + ldr) * K + ldc;
    __bf16* sA0 = &As[0][ldr * 32 + ldc];
    __bf16* sB0 = &Bs[0][ldr * 32 + ldc];
    __bf16* sA1 = &As[1][ldr * 32 + ldc];
    __bf16* sB1 = &Bs[1][ldr * 32 + ldc];

    // ---- preload K-tile 0 into buffer 0 ----
#if USE_ASYNC
    ASYNC_CP16(gA,     sA0);
    ASYNC_CP16(gA + 8, sA0 + 8);
    ASYNC_CP16(gB,     sB0);
    ASYNC_CP16(gB + 8, sB0 + 8);
    ASYNC_WAIT();
#else
    *(v8bf*)sA0       = *(const v8bf*)(gA);
    *(v8bf*)(sA0 + 8) = *(const v8bf*)(gA + 8);
    *(v8bf*)sB0       = *(const v8bf*)(gB);
    *(v8bf*)(sB0 + 8) = *(const v8bf*)(gB + 8);
#endif
    __syncthreads();

    const int nk = K >> 5;
    int cur = 0;
    for (int kt = 0; kt < nk; ++kt) {
        const int k0 = kt << 5;
        const bool more = (kt + 1 < nk);
        __bf16* dA = cur ? sA0 : sA1;   // next buffer
        __bf16* dB = cur ? sB0 : sB1;

        // ---- stage next K-tile into the other buffer (overlaps the WMMA chain) ----
#if USE_ASYNC
        if (more) {
            ASYNC_CP16(gA + k0 + 32, dA);
            ASYNC_CP16(gA + k0 + 40, dA + 8);
            ASYNC_CP16(gB + k0 + 32, dB);
            ASYNC_CP16(gB + k0 + 40, dB + 8);
        }
#else
        v8bf pa0 = {}, pa1 = {}, pb0 = {}, pb1 = {};
        if (more) {
            pa0 = *(const v8bf*)(gA + k0 + 32);
            pa1 = *(const v8bf*)(gA + k0 + 40);
            pb0 = *(const v8bf*)(gB + k0 + 32);
            pb1 = *(const v8bf*)(gB + k0 + 40);
        }
#endif

        // ---- compute from current buffer: batch all fragment loads, then 8 WMMAs ----
        const v8bf* arow = (const v8bf*)&As[cur][(wave * 16 + l16) * 32];
        v16bf af = SHUF16(arow[kch], arow[kch + 2]);
        v16bf bfr[8];
#pragma unroll
        for (int nt = 0; nt < 8; ++nt) {
            const v8bf* brow = (const v8bf*)&Bs[cur][(nt * 16 + l16) * 32];
            bfr[nt] = SHUF16(brow[kch], brow[kch + 2]);
        }
#pragma unroll
        for (int nt = 0; nt < 8; ++nt) {
            acc[nt] = __builtin_amdgcn_wmma_f32_16x16x32_bf16(false, af, false, bfr[nt],
                                                              (short)0, acc[nt], false, false);
        }
        // scheduling: all 18 DS reads (A:2 + B:16) first, then the 8 WMMAs
        SGB(0x100, 18);
        SGB(0x008, 8);

#if USE_ASYNC
        if (more) ASYNC_WAIT();
#else
        if (more) {
            *(v8bf*)dA       = pa0;
            *(v8bf*)(dA + 8) = pa1;
            *(v8bf*)dB       = pb0;
            *(v8bf*)(dB + 8) = pb1;
        }
#endif
        __syncthreads();
        cur ^= 1;
    }

    // epilogue: C layout row = r + 8*(lane>=16), col = lane%16
    const int mrow0 = mbase + wave * 16 + ((lane >> 4) << 3);
#pragma unroll
    for (int nt = 0; nt < 8; ++nt) {
        int g = nbase + nt * 16 + l16;
        float bv = bias[g];
#pragma unroll
        for (int r = 0; r < 8; ++r) {
            int m = mrow0 + r;
            int t = m & (T_ - 1);
            int b = m >> 9;  // /512
            xg[(size_t)(t * B_ + b) * G3H + g] = (__bf16)(acc[nt][r] + bv);
        }
    }
}

// ---------------- grid-wide sense-reversing barrier ----------------
__device__ __forceinline__ void grid_sync(volatile unsigned* cnt, volatile unsigned* gen, unsigned nb) {
    __threadfence();
    __syncthreads();
    if (threadIdx.x == 0) {
        unsigned g = *gen;
        if (atomicAdd((unsigned*)cnt, 1u) == nb - 1u) {
            *cnt = 0u;
            __threadfence();
            atomicAdd((unsigned*)gen, 1u);
        } else {
            while (*gen == g) { __builtin_amdgcn_s_sleep(2); }
        }
    }
    __syncthreads();
}

// ---------------- persistent GRU scan ----------------
// 64 blocks x 128 threads. Block owns hidden slice [j0, j0+16).
// Its 96KB w_hh slice (3 gates x 16 rows x 1024 K, bf16) stays in LDS for all T steps.
__global__ __launch_bounds__(128) void gru_rec(const __bf16* __restrict__ xg,   // [T*B][3H]
                                               const __bf16* __restrict__ whh,  // [3H][H]
                                               const float* __restrict__ bhh,   // [3H]
                                               float* h32, __bf16* hbf,         // [2][B][H] each
                                               __bf16* seq_out,                 // [B][T][H] or null
                                               float* final_out,                // [B][H] or null
                                               unsigned* bar, int nblocks) {
    extern __shared__ __align__(16) char smem_raw[];
    __bf16* wsm = (__bf16*)smem_raw;  // [48][H_]  rows: gate*16 + n
    const int tid  = threadIdx.x;
    const int lane = tid & 31;
    const int wave = tid >> 5;   // m-tile 0..3 (batch rows wave*16 .. wave*16+15)
    const int l16  = lane & 15;
    const int kch  = lane >> 4;
    const int j0   = blockIdx.x * 16;

    // Stage weight slice into LDS once (resident across all 512 steps).
    for (int idx = tid; idx < 48 * (H_ / 8); idx += 128) {
        int row = idx / (H_ / 8);
        int ch  = idx % (H_ / 8);
        int g = row >> 4, n = row & 15;
        const __bf16* src = &whh[(size_t)(g * H_ + j0 + n) * H_ + ch * 8];
#if USE_ASYNC
        ASYNC_CP16(src, &wsm[idx * 8]);
#else
        ((v8bf*)wsm)[idx] = *(const v8bf*)src;
#endif
    }
#if USE_ASYNC
    ASYNC_WAIT();
#endif
    // Zero this block's hidden slice in both h buffers (h0 = 0).
    for (int idx = tid; idx < B_ * 16; idx += 128) {
        int b = idx >> 4, c = idx & 15;
        h32[(size_t)b * H_ + j0 + c]        = 0.f;
        h32[(size_t)(B_ + b) * H_ + j0 + c] = 0.f;
        hbf[(size_t)b * H_ + j0 + c]        = (__bf16)0.f;
        hbf[(size_t)(B_ + b) * H_ + j0 + c] = (__bf16)0.f;
    }
    grid_sync(bar, bar + 1, (unsigned)nblocks);

    const float bhr = bhh[0 * H_ + j0 + l16];
    const float bhz = bhh[1 * H_ + j0 + l16];
    const float bhn = bhh[2 * H_ + j0 + l16];

    int cur = 0;
    for (int t = 0; t < T_; ++t) {
        // Pull next step's xg rows toward the caches while this step computes.
        if (t + 1 < T_) {
            const __bf16* nx = xg + (size_t)(t + 1) * B_ * G3H +
                               (size_t)(wave * 16 + l16) * G3H + j0;
            __builtin_prefetch((const void*)nx, 0, 1);
        }

        const __bf16* hb = hbf + (size_t)cur * B_ * H_;
        v8f aR = {0.f, 0.f, 0.f, 0.f, 0.f, 0.f, 0.f, 0.f};
        v8f aZ = aR, aN = aR;

#pragma unroll 4
        for (int k0 = 0; k0 < H_; k0 += 32) {
            const v8bf* ar = (const v8bf*)&hb[(size_t)(wave * 16 + l16) * H_ + k0];
            v16bf af = SHUF16(ar[kch], ar[kch + 2]);
            const v8bf* br0 = (const v8bf*)&wsm[(0 * 16 + l16) * H_ + k0];
            const v8bf* br1 = (const v8bf*)&wsm[(1 * 16 + l16) * H_ + k0];
            const v8bf* br2 = (const v8bf*)&wsm[(2 * 16 + l16) * H_ + k0];
            v16bf bR = SHUF16(br0[kch], br0[kch + 2]);
            v16bf bZ = SHUF16(br1[kch], br1[kch + 2]);
            v16bf bN = SHUF16(br2[kch], br2[kch + 2]);
            aR = __builtin_amdgcn_wmma_f32_16x16x32_bf16(false, af, false, bR, (short)0, aR, false, false);
            aZ = __builtin_amdgcn_wmma_f32_16x16x32_bf16(false, af, false, bZ, (short)0, aZ, false, false);
            aN = __builtin_amdgcn_wmma_f32_16x16x32_bf16(false, af, false, bN, (short)0, aN, false, false);
            // scheduling: A (2 global reads) + B (6 DS reads) first, then 3 WMMAs
            SGB(0x020, 2);
            SGB(0x100, 6);
            SGB(0x008, 3);
        }

        const float* hr = h32 + (size_t)cur * B_ * H_;
        float* hw   = h32 + (size_t)(cur ^ 1) * B_ * H_;
        __bf16* hbw = hbf + (size_t)(cur ^ 1) * B_ * H_;
        const size_t xbase = (size_t)t * B_ * G3H;
        const int j = j0 + l16;
#pragma unroll
        for (int r = 0; r < 8; ++r) {
            int b = wave * 16 + ((lane >> 4) << 3) + r;
            const __bf16* xrow = xg + xbase + (size_t)b * G3H;
            float xr = (float)xrow[j];
            float xz = (float)xrow[H_ + j];
            float xn = (float)xrow[2 * H_ + j];
            float rr = sigmoid_f(xr + aR[r] + bhr);
            float zz = sigmoid_f(xz + aZ[r] + bhz);
            float nn = tanhf(xn + rr * (aN[r] + bhn));
            float hold = hr[(size_t)b * H_ + j];
            float hnew = (1.f - zz) * nn + zz * hold;
            hw[(size_t)b * H_ + j]  = hnew;
            hbw[(size_t)b * H_ + j] = (__bf16)hnew;
            if (seq_out)   seq_out[((size_t)b * T_ + t) * H_ + j] = (__bf16)hnew;
            if (final_out && t == T_ - 1) final_out[(size_t)b * H_ + j] = hnew;
        }
        grid_sync(bar, bar + 1, (unsigned)nblocks);
        cur ^= 1;
    }
}

// ---------------- orchestration ----------------
extern "C" void kernel_launch(void* const* d_in, const int* in_sizes, int n_in,
                              void* d_out, int out_size, void* d_ws, size_t ws_size,
                              hipStream_t stream) {
    (void)in_sizes; (void)n_in; (void)out_size; (void)ws_size;
    const float* x     = (const float*)d_in[0];
    const float* w_ih0 = (const float*)d_in[1];
    const float* w_hh0 = (const float*)d_in[2];
    const float* b_ih0 = (const float*)d_in[3];
    const float* b_hh0 = (const float*)d_in[4];
    const float* w_ih1 = (const float*)d_in[5];
    const float* w_hh1 = (const float*)d_in[6];
    const float* b_ih1 = (const float*)d_in[7];
    const float* b_hh1 = (const float*)d_in[8];
    float* out = (float*)d_out;

    char* p = (char*)d_ws;
    auto carve = [&](size_t bytes) -> char* {
        char* r = p;
        p += (bytes + 255) & ~(size_t)255;
        return r;
    };
    unsigned* bar = (unsigned*)carve(256);
    __bf16* xbuf0 = (__bf16*)carve((size_t)B_ * T_ * I_ * 2);      // layer-0 input, bf16
    __bf16* xbuf1 = (__bf16*)carve((size_t)B_ * T_ * H_ * 2);      // layer-0 outputs = layer-1 input
    __bf16* wih   = (__bf16*)carve((size_t)G3H * H_ * 2);
    __bf16* whh   = (__bf16*)carve((size_t)G3H * H_ * 2);
    __bf16* xg    = (__bf16*)carve((size_t)B_ * T_ * G3H * 2);     // [t][b][3H]
    float*  h32   = (float*)carve((size_t)2 * B_ * H_ * 4);
    __bf16* hbf   = (__bf16*)carve((size_t)2 * B_ * H_ * 2);

    init_barrier<<<1, 1, 0, stream>>>(bar);

    const size_t rec_lds = (size_t)48 * H_ * 2;  // 96 KB dynamic LDS

    // ---- Layer 0 ----
    cvt_f32_bf16<<<1024, 256, 0, stream>>>(x, xbuf0, (long long)B_ * T_ * I_);
    cvt_f32_bf16<<<1024, 256, 0, stream>>>(w_ih0, wih, (long long)G3H * I_);
    cvt_f32_bf16<<<1024, 256, 0, stream>>>(w_hh0, whh, (long long)G3H * H_);
    gemm_xg<<<dim3(G3H / 128, (B_ * T_) / 128), 256, 0, stream>>>(xbuf0, wih, b_ih0, xg, I_);
    gru_rec<<<dim3(H_ / 16), dim3(128), rec_lds, stream>>>(xg, whh, b_hh0, h32, hbf,
                                                           xbuf1, nullptr, bar, H_ / 16);

    // ---- Layer 1 ----
    cvt_f32_bf16<<<1024, 256, 0, stream>>>(w_ih1, wih, (long long)G3H * H_);
    cvt_f32_bf16<<<1024, 256, 0, stream>>>(w_hh1, whh, (long long)G3H * H_);
    gemm_xg<<<dim3(G3H / 128, (B_ * T_) / 128), 256, 0, stream>>>(xbuf1, wih, b_ih1, xg, H_);
    gru_rec<<<dim3(H_ / 16), dim3(128), rec_lds, stream>>>(xg, whh, b_hh1, h32, hbf,
                                                           nullptr, out, bar, H_ / 16);
}
